// MultiHeadAttention_51419348468290
// MI455X (gfx1250) — matrix operand order repaired and checked by reading, not compiled, not run on hardware
//
#include <hip/hip_runtime.h>
#include <hip/hip_bf16.h>

// ---------------------------------------------------------------------------
// Multi-head attention (B=4, S=2048, d_model=1024, H=16, d_k=64) + out-proj.
// Pipeline:
//   0) one bandwidth-bound pass converts Q(*1/sqrt(dk)), K, V, W_out -> f16
//   1) flash attention, transposed score tiles (S^T = K*Q^T) so softmax is
//      in-lane; K tile staged to LDS via CDNA5 async global->LDS copies;
//      V tile staged transposed by register re-packing; WMMA f16->f32.
//   2) projection GEMM on pure f16 operands (no conversions in the loop).
// ---------------------------------------------------------------------------

#define BATCH   4
#define S_LEN   2048
#define DMODEL  1024
#define NH      16
#define DK      64
#define SCALE   0.125f   /* 1/sqrt(64) */

typedef __attribute__((ext_vector_type(16))) _Float16 v16h;
typedef __attribute__((ext_vector_type(8)))  _Float16 v8h;
typedef __attribute__((ext_vector_type(2)))  _Float16 v2h;
typedef __attribute__((ext_vector_type(8)))  float    v8f;

// Load 8 consecutive halves (16B-aligned, LDS or global) into half of a v16h.
__device__ __forceinline__ void loadh8(const _Float16* s, v16h& d, const int base) {
    v8h t = *reinterpret_cast<const v8h*>(s);
#pragma unroll
    for (int i = 0; i < 8; ++i) d[base + i] = t[i];
}

// Convert 8 floats -> v8h and store as one 16B write.
__device__ __forceinline__ void storeh8(_Float16* dst, const float* e) {
    v8h t;
#pragma unroll
    for (int i = 0; i < 8; ++i) t[i] = (_Float16)e[i];
    *reinterpret_cast<v8h*>(dst) = t;
}

__device__ __forceinline__ v8f wmma_f16(v16h a, v16h b, v8f c) {
    return __builtin_amdgcn_wmma_f32_16x16x32_f16(
        /*neg_a=*/false, a, /*neg_b=*/false, b,
        /*c_mod=*/(short)0, c, /*reuse_a=*/false, /*reuse_b=*/false);
}

// ---------------------------------------------------------------------------
// Kernel 0: f32 -> f16 conversion with optional scale (vectorized, 8/thread).
// ---------------------------------------------------------------------------
__global__ __launch_bounds__(256)
void cvt_f16_kernel(const float* __restrict__ src, _Float16* __restrict__ dst,
                    const float scale) {
    const size_t i = ((size_t)blockIdx.x * 256 + threadIdx.x) * 8;
    const float4* s4 = reinterpret_cast<const float4*>(src + i);
    float4 a = s4[0];
    float4 b = s4[1];
    float t[8] = {a.x * scale, a.y * scale, a.z * scale, a.w * scale,
                  b.x * scale, b.y * scale, b.z * scale, b.w * scale};
    storeh8(dst + i, t);
}

// ---------------------------------------------------------------------------
// Kernel 1: attention.  Block = 128 threads (4 waves), one (b,h) pair and
// 64 query rows; each wave owns a 16-query tile.  All operands f16.
// grid.x = B * NH * (S/64) = 2048
// ---------------------------------------------------------------------------
__global__ __launch_bounds__(128)
void mha_attn_kernel(const _Float16* __restrict__ qh,   /* pre-scaled */
                     const _Float16* __restrict__ kh,
                     const _Float16* __restrict__ vh,
                     _Float16* __restrict__ attn_out /* [B,S,DMODEL] f16 */) {
    // KH: K tile row-major f16 [key][d]  -> score-A reads are contiguous 16B.
    // VT: V tile transposed  f16 [d][key] -> PV-A reads are contiguous 16B.
    // PT: per-wave P^T, query-major [query][key] -> B reads contiguous 16B.
    __shared__ __align__(32) _Float16 KH[32][72];        // 144B stride
    __shared__ __align__(32) _Float16 VT[DK][40];        // 80B stride
    __shared__ __align__(32) _Float16 PT[4][16][40];

    const int wave = threadIdx.x >> 5;
    const int lane = threadIdx.x & 31;
    const int lh   = lane >> 4;    // lane half (0/1)
    const int ln   = lane & 15;

    const int qt = blockIdx.x & 31;          // S/64 = 32 query tiles
    const int bh = blockIdx.x >> 5;
    const int h  = bh & (NH - 1);
    const int b  = bh >> 4;

    const int q0 = qt * 64 + wave * 16;

    const _Float16* qbase = qh + (size_t)b * S_LEN * DMODEL + (size_t)h * DK;
    const _Float16* kbase = kh + (size_t)b * S_LEN * DMODEL + (size_t)h * DK;
    const _Float16* vbase = vh + (size_t)b * S_LEN * DMODEL + (size_t)h * DK;

    // ---- B-layout Q^T tile (already scaled), direct f16 loads ----
    const _Float16* qrow = qbase + (size_t)(q0 + ln) * DMODEL;
    v16h Bq0, Bq1;
    loadh8(qrow +  0 + 8 * lh, Bq0, 0);
    loadh8(qrow + 16 + 8 * lh, Bq0, 8);
    loadh8(qrow + 32 + 8 * lh, Bq1, 0);
    loadh8(qrow + 48 + 8 * lh, Bq1, 8);

    // O^T accumulators: four 16x16 C tiles; VGPR r = dim (16t + 8*lh + r),
    // lane = query (ln).  Per-lane softmax state (query-local).
    v8f o0 = {}, o1 = {}, o2 = {}, o3 = {};
    float m_run = -1e30f, l_run = 0.0f;

    // Staging coordinates (loop-invariant).
    const int rr = threadIdx.x & 31;           // K: key row
    const int cc = (threadIdx.x >> 5) * 16;    // K: 16-half dim slice
    const int kk = (threadIdx.x & 15) * 2;     // V: pair of adjacent keys
    const int c2 = (threadIdx.x >> 4) * 8;     // V: 8-dim slice

    for (int j0 = 0; j0 < S_LEN; j0 += 32) {
        __syncthreads();
        // ---- K tile: CDNA5 async global->LDS copy (2 x b128 per thread) ----
        {
            const _Float16* krow = kbase + (size_t)(j0 + rr) * DMODEL + cc;
            unsigned lds0 = (unsigned)(size_t)(&KH[rr][cc]);
            unsigned lds1 = lds0 + 16u;
            asm volatile("global_load_async_to_lds_b128 %0, %1, off"
                         :: "v"(lds0), "v"(krow) : "memory");
            asm volatile("global_load_async_to_lds_b128 %0, %1, off"
                         :: "v"(lds1), "v"(krow + 8) : "memory");
            if (j0 + 32 < S_LEN) __builtin_prefetch(krow + (size_t)32 * DMODEL, 0, 0);
        }
        // ---- V tile: transpose by register re-packing (no conversions) ----
        {
            const _Float16* v0r = vbase + (size_t)(j0 + kk) * DMODEL + c2;
            const _Float16* v1r = v0r + DMODEL;
            v8h va = *reinterpret_cast<const v8h*>(v0r);
            v8h vb = *reinterpret_cast<const v8h*>(v1r);
#pragma unroll
            for (int i = 0; i < 8; ++i) {
                v2h p;
                p[0] = va[i];
                p[1] = vb[i];
                *reinterpret_cast<v2h*>(&VT[c2 + i][kk]) = p;
            }
            if (j0 + 32 < S_LEN) __builtin_prefetch(v0r + (size_t)32 * DMODEL, 0, 0);
        }
        asm volatile("s_wait_asynccnt 0x0" ::: "memory");
        __syncthreads();

        // ---- S^T = K (16x64) * Q^T (64x16), two 16-key groups, 4 WMMAs ----
        v8f c0 = {}, c1 = {};
        {
            v16h Ak;
            loadh8(&KH[ 0 + ln][ 0 + 8 * lh], Ak, 0); loadh8(&KH[ 0 + ln][16 + 8 * lh], Ak, 8);
            c0 = wmma_f16(Ak, Bq0, c0);
            loadh8(&KH[ 0 + ln][32 + 8 * lh], Ak, 0); loadh8(&KH[ 0 + ln][48 + 8 * lh], Ak, 8);
            c0 = wmma_f16(Ak, Bq1, c0);
            loadh8(&KH[16 + ln][ 0 + 8 * lh], Ak, 0); loadh8(&KH[16 + ln][16 + 8 * lh], Ak, 8);
            c1 = wmma_f16(Ak, Bq0, c1);
            loadh8(&KH[16 + ln][32 + 8 * lh], Ak, 0); loadh8(&KH[16 + ln][48 + 8 * lh], Ak, 8);
            c1 = wmma_f16(Ak, Bq1, c1);
        }

        // ---- Online softmax: per-lane (per-query) stats, in-register ----
        float mt = c0[0];
#pragma unroll
        for (int r = 1; r < 8; ++r) mt = fmaxf(mt, c0[r]);
#pragma unroll
        for (int r = 0; r < 8; ++r) mt = fmaxf(mt, c1[r]);
        mt = fmaxf(mt, __shfl_xor(mt, 16, 32));   // merge the two key halves

        const float mnew  = fmaxf(m_run, mt);
        const float alpha = __expf(m_run - mnew);
        m_run = mnew;

        float e0[8], e1[8];
        float ls = 0.0f;
#pragma unroll
        for (int r = 0; r < 8; ++r) {
            e0[r] = __expf(c0[r] - mnew);
            e1[r] = __expf(c1[r] - mnew);
            ls += e0[r] + e1[r];
        }
        ls += __shfl_xor(ls, 16, 32);
        l_run = l_run * alpha + ls;

#pragma unroll
        for (int r = 0; r < 8; ++r) {
            o0[r] *= alpha; o1[r] *= alpha; o2[r] *= alpha; o3[r] *= alpha;
        }

        // P^T spill: consecutive keys per lane -> two vector LDS stores.
        storeh8(&PT[wave][ln][ 0 + 8 * lh], e0);
        storeh8(&PT[wave][ln][16 + 8 * lh], e1);

        // ---- O^T += V^T (64x32) * P^T (32x16) via 4 WMMAs, shared B ----
        v16h Bp;
        loadh8(&PT[wave][ln][8 * lh],      Bp, 0);
        loadh8(&PT[wave][ln][16 + 8 * lh], Bp, 8);
        {
            v16h Av;
            loadh8(&VT[ 0 + ln][8 * lh], Av, 0); loadh8(&VT[ 0 + ln][16 + 8 * lh], Av, 8);
            o0 = wmma_f16(Av, Bp, o0);
            loadh8(&VT[16 + ln][8 * lh], Av, 0); loadh8(&VT[16 + ln][16 + 8 * lh], Av, 8);
            o1 = wmma_f16(Av, Bp, o1);
            loadh8(&VT[32 + ln][8 * lh], Av, 0); loadh8(&VT[32 + ln][16 + 8 * lh], Av, 8);
            o2 = wmma_f16(Av, Bp, o2);
            loadh8(&VT[48 + ln][8 * lh], Av, 0); loadh8(&VT[48 + ln][16 + 8 * lh], Av, 8);
            o3 = wmma_f16(Av, Bp, o3);
        }
    }

    // ---- Normalize and store f16: per lane dims contiguous -> 16B stores ----
    const float inv = 1.0f / l_run;
    _Float16* orow = attn_out + (size_t)b * S_LEN * DMODEL
                   + (size_t)(q0 + ln) * DMODEL + (size_t)h * DK;
    {
        v8f oo[4] = {o0, o1, o2, o3};
#pragma unroll
        for (int t = 0; t < 4; ++t) {
            float e[8];
#pragma unroll
            for (int r = 0; r < 8; ++r) e[r] = oo[t][r] * inv;
            storeh8(orow + 16 * t + 8 * lh, e);
        }
    }
}

// ---------------------------------------------------------------------------
// Kernel 2: output projection  out[i][o] = sum_k X[i][k]*W[o][k] + bias[o].
// X and W already f16 -> inner loop is 4 x v8h loads + 1 WMMA.
// Block = 128 threads (4 waves); wave = 16x16 output tile.
// grid.x = (8192/16) * (1024/64) = 8192
// ---------------------------------------------------------------------------
__global__ __launch_bounds__(128)
void mha_proj_kernel(const _Float16* __restrict__ X,
                     const _Float16* __restrict__ W,
                     const float* __restrict__ bias,
                     float* __restrict__ out) {
    const int wave = threadIdx.x >> 5;
    const int lane = threadIdx.x & 31;
    const int lh   = lane >> 4;
    const int ln   = lane & 15;

    const int cb = blockIdx.x & 15;          // 16 column blocks of 64
    const int rb = blockIdx.x >> 4;          // 512 row blocks of 16
    const int i0 = rb * 16;
    const int o0 = cb * 64 + wave * 16;

    const _Float16* xrow = X + (size_t)(i0 + ln) * DMODEL;
    const _Float16* wrow = W + (size_t)(o0 + ln) * DMODEL;

    v8f acc = {};
    for (int kc = 0; kc < DMODEL; kc += 32) {
        v16h Ax, Bw;
        loadh8(xrow + kc +  0 + 8 * lh, Ax, 0);
        loadh8(xrow + kc + 16 + 8 * lh, Ax, 8);
        loadh8(wrow + kc +  0 + 8 * lh, Bw, 0);
        loadh8(wrow + kc + 16 + 8 * lh, Bw, 8);
        acc = wmma_f16(Ax, Bw, acc);
    }

    const float bval = bias[o0 + ln];
#pragma unroll
    for (int r = 0; r < 8; ++r) {
        out[(size_t)(i0 + r + 8 * lh) * DMODEL + o0 + ln] = acc[r] + bval;
    }
}

// ---------------------------------------------------------------------------
extern "C" void kernel_launch(void* const* d_in, const int* in_sizes, int n_in,
                              void* d_out, int out_size, void* d_ws, size_t ws_size,
                              hipStream_t stream) {
    const float* q    = (const float*)d_in[0];
    const float* k    = (const float*)d_in[1];
    const float* v    = (const float*)d_in[2];
    const float* Wout = (const float*)d_in[3];
    const float* bout = (const float*)d_in[4];
    float* out = (float*)d_out;

    const size_t NQKV = (size_t)BATCH * S_LEN * DMODEL;   // 8,388,608
    const size_t NW   = (size_t)DMODEL * DMODEL;          // 1,048,576

    _Float16* qhb = (_Float16*)d_ws;
    _Float16* khb = qhb + NQKV;
    _Float16* vhb = khb + NQKV;
    _Float16* whb = vhb + NQKV;
    _Float16* xhb = whb + NW;      // attention output, f16

    dim3 blk256(256);
    cvt_f16_kernel<<<dim3(NQKV / (256 * 8)), blk256, 0, stream>>>(q, qhb, SCALE);
    cvt_f16_kernel<<<dim3(NQKV / (256 * 8)), blk256, 0, stream>>>(k, khb, 1.0f);
    cvt_f16_kernel<<<dim3(NQKV / (256 * 8)), blk256, 0, stream>>>(v, vhb, 1.0f);
    cvt_f16_kernel<<<dim3(NW   / (256 * 8)), blk256, 0, stream>>>(Wout, whb, 1.0f);

    dim3 blk128(128);
    dim3 grid1(BATCH * NH * (S_LEN / 64));                     // 2048 blocks
    mha_attn_kernel<<<grid1, blk128, 0, stream>>>(qhb, khb, vhb, xhb);

    dim3 grid2((BATCH * S_LEN / 16) * (DMODEL / 64));          // 8192 blocks
    mha_proj_kernel<<<grid2, blk128, 0, stream>>>(xhb, whb, bout, out);
}
